// _HFSparseMoeBlock_5162550689806
// MI455X (gfx1250) — matrix operand order repaired and checked by reading, not compiled, not run on hardware
//
#include <hip/hip_runtime.h>
#include <hip/hip_bf16.h>
#include <math.h>
#include <stdint.h>

// ---------------- problem constants (fixed by reference setup) ----------------
#define HD   2048   // hidden size H
#define ID   1024   // intermediate size I
#define NE   8      // experts
#define NT   4096   // tokens T = B*S
#define MT   32     // token tile per block
#define KC   64     // K-chunk staged in LDS (phase 1, K = H)
#define IT   128    // i-slice width (phase 1 output tile)
#define NC   128    // h-chunk width (phase 2 output tile)

typedef __attribute__((ext_vector_type(16))) __bf16   v16bf;
typedef __attribute__((ext_vector_type(8)))  float    v8f;
typedef __attribute__((ext_vector_type(8)))  unsigned v8u;

// fp32 -> bf16, round-to-nearest-even
static __device__ inline unsigned short f2bf(float f) {
  unsigned u = __float_as_uint(f);
  u += 0x7fffu + ((u >> 16) & 1u);
  return (unsigned short)(u >> 16);
}
static __device__ inline unsigned pk2(float lo, float hi) {
  return (unsigned)f2bf(lo) | ((unsigned)f2bf(hi) << 16);
}

static __device__ inline v8f wmma_bf16(v16bf a, v16bf b, v8f c) {
  // D = A(16x32 bf16) * B(32x16 bf16) + C(16x16 f32)
  return __builtin_amdgcn_wmma_f32_16x16x32_bf16(false, a, false, b, (short)0, c,
                                                 false, false);
}

// Async global->LDS copy of 16 bytes (GLOBAL_LOAD_ASYNC_TO_LDS_B128, ASYNCcnt).
// lds_addr: LDS byte address (low 32 bits of flat addr of a shared pointer).
static __device__ inline void async_copy_b128(unsigned lds_addr,
                                              const void* gaddr) {
  asm volatile("global_load_async_to_lds_b128 %0, %1, off"
               :
               : "v"(lds_addr), "v"((unsigned long long)(uintptr_t)gaddr)
               : "memory");
}
static __device__ inline void wait_asynccnt0() {
  asm volatile("s_wait_asynccnt 0x0" ::: "memory");
}

// A-operand gather from a row-major bf16 LDS tile (row stride = 2*SWH bf16).
// A in contiguous k order (same as ldsB): lane L row M=L%16; VGPR v: K=16*(L/16)+2v(+ks).
// Two bf16 per dword.
template <int SWH>
static __device__ inline v16bf ldsA(const unsigned* __restrict__ Lu, int m,
                                    int ks, int khalf) {
  v8u u;
  const int r = m * SWH + (ks >> 1) + (khalf << 3);
#pragma unroll
  for (int v = 0; v < 4; ++v) u[v] = Lu[r + v];
#pragma unroll
  for (int v = 0; v < 4; ++v) u[4 + v] = Lu[r + 4 + v];
  union { v8u a; v16bf b; } c; c.a = u; return c.b;
}

// B-operand gather: lane L col N=L%16; VGPR v: K = 16*(L/16) + 2v (+ks).
template <int SWH>
static __device__ inline v16bf ldsB(const unsigned* __restrict__ Lu, int n,
                                    int ks, int khalf) {
  v8u u;
  const int r = n * SWH + (ks >> 1) + (khalf << 3);
#pragma unroll
  for (int v = 0; v < 8; ++v) u[v] = Lu[r + v];
  union { v8u a; v16bf b; } c; c.a = u; return c.b;
}

// ---------------- kernel 0: zero output + bucket counts ----------------
__global__ void __launch_bounds__(256) moe_init_kernel(float* __restrict__ out,
                                                       int n,
                                                       int* __restrict__ counts) {
  const int i = blockIdx.x * 256 + threadIdx.x;
  if (i < n) out[i] = 0.0f;
  if (i < NE) counts[i] = 0;
}

// ---------------- kernel 1: router (+ x -> bf16 conversion) ----------------
__global__ void __launch_bounds__(256) moe_router_kernel(
    const float* __restrict__ x, const float* __restrict__ gw,
    const float* __restrict__ bias, unsigned short* __restrict__ xb,
    float* __restrict__ combine, int* __restrict__ counts,
    int* __restrict__ lists) {
  const int t = blockIdx.x;
  const int tid = threadIdx.x;
  __shared__ float red[256][NE];

  float p[NE];
#pragma unroll
  for (int e = 0; e < NE; ++e) p[e] = 0.0f;

  for (int h = tid; h < HD; h += 256) {
    const float xv = x[(size_t)t * HD + h];
    xb[(size_t)t * HD + h] = f2bf(xv);
#pragma unroll
    for (int e = 0; e < NE; ++e) p[e] += xv * gw[e * HD + h];
  }
#pragma unroll
  for (int e = 0; e < NE; ++e) red[tid][e] = p[e];
  __syncthreads();
  for (int s = 128; s > 0; s >>= 1) {
    if (tid < s) {
#pragma unroll
      for (int e = 0; e < NE; ++e) red[tid][e] += red[tid + s][e];
    }
    __syncthreads();
  }
  if (tid == 0) {
    float sig[NE], sc[NE];
#pragma unroll
    for (int e = 0; e < NE; ++e) {
      sig[e] = 1.0f / (1.0f + __expf(-red[0][e]));
      sc[e] = sig[e] + bias[e];
    }
    int b0 = 0;
    for (int e = 1; e < NE; ++e) if (sc[e] > sc[b0]) b0 = e;
    int b1 = (b0 == 0) ? 1 : 0;
    for (int e = 0; e < NE; ++e) if (e != b0 && sc[e] > sc[b1]) b1 = e;
    const float wsum = sig[b0] + sig[b1];
#pragma unroll
    for (int e = 0; e < NE; ++e) combine[t * NE + e] = 0.0f;
    combine[t * NE + b0] = sig[b0] / wsum;
    combine[t * NE + b1] = sig[b1] / wsum;
    int s0 = atomicAdd(&counts[b0], 1); lists[b0 * NT + s0] = t;
    int s1 = atomicAdd(&counts[b1], 1); lists[b1 * NT + s1] = t;
  }
}

// ---------------- kernel 2: fused grouped SwiGLU expert GEMM ----------------
// grid = (NE, NT/MT); block = 256 threads (8 wave32).
// Dynamic LDS carve (100 KB, fits the 320 KB WGP LDS):
//   [0,      65536)  ACT  : bf16 [MT=32][ID=1024]   (persists both phases)
//   [65536,  69632)  X    : bf16 [32][KC=64]        (phase 1, async-copied)
//   [69632,  86016)  W1   : bf16 [IT=128][KC=64]    (phase 1)
//   [86016, 102400)  W3   : bf16 [IT=128][KC=64]    (phase 1)
//   [65536,  98304)  W2   : bf16 [NC=128][128]      (phase 2, reuses X/W1 space)
__global__ void __launch_bounds__(256) moe_expert_kernel(
    const float* __restrict__ w1, const float* __restrict__ w3,
    const float* __restrict__ w2, const unsigned short* __restrict__ xb,
    const float* __restrict__ combine, const int* __restrict__ counts,
    const int* __restrict__ lists, float* __restrict__ out) {
  const int e = blockIdx.x;
  const int mbase = blockIdx.y * MT;
  const int cnt = counts[e];
  if (mbase >= cnt) return;  // uniform per block

  extern __shared__ char smem[];
  unsigned short* ACT = (unsigned short*)smem;
  unsigned* ACTu = (unsigned*)smem;
  unsigned short* Xl = (unsigned short*)(smem + 65536);
  unsigned* Xu  = (unsigned*)(smem + 65536);
  unsigned* W1u = (unsigned*)(smem + 69632);
  unsigned* W3u = (unsigned*)(smem + 86016);
  unsigned* W2u = (unsigned*)(smem + 65536);

  __shared__ int   s_idx[MT];
  __shared__ float s_cw[MT];

  const int tid = threadIdx.x;
  if (tid < MT) {
    int t = -1; float cw = 0.0f;
    if (mbase + tid < cnt) {
      t = lists[e * NT + mbase + tid];
      cw = combine[t * NE + e];
    }
    s_idx[tid] = t; s_cw[tid] = cw;
  }
  __syncthreads();

  const int wave  = tid >> 5;
  const int lane  = tid & 31;
  const int khalf = lane >> 4;
  const int l16   = lane & 15;
  const int mt = (wave & 1) * 16;    // 16-row sub-tile within 32 tokens
  const int ng = (wave >> 1) * 32;   // 32-col group within 128 (2 n-tiles/wave)

  const v8f zero8 = {0.f, 0.f, 0.f, 0.f, 0.f, 0.f, 0.f, 0.f};

  // X-tile async-copy geometry: thread -> (row, 8-elem col group), 16B each.
  const int xr = tid >> 3;
  const int xc = (tid & 7) * 8;
  // Padded rows (idx<0) point at token 0: harmless garbage, their D-rows are
  // never stored (WMMA row m of D depends only on row m of A).
  const int xtok = (s_idx[xr] < 0) ? 0 : s_idx[xr];
  const unsigned x_lds = (unsigned)(uintptr_t)(Xl + (xr * KC + xc));

  // ---------------- Phase 1: h1/h3 = x @ w1^T / w3^T, act = silu(h1)*h3 ----
  for (int isl = 0; isl < ID; isl += IT) {
    v8f a1[2], a3[2];
    a1[0] = zero8; a1[1] = zero8; a3[0] = zero8; a3[1] = zero8;

    for (int kk = 0; kk < HD; kk += KC) {
      // stage X: pure bf16 byte-copy -> ASYNCcnt path (no VGPR round trip)
      async_copy_b128(x_lds, xb + (size_t)xtok * HD + kk + xc);

      {  // stage w1/w3: 128 x 64 fp32 -> bf16 (VALU convert in flight)
        const int r = tid >> 1, c0 = (tid & 1) * 32;
        const size_t base = ((size_t)e * ID + isl + r) * HD + kk + c0;
        const float4* s1 = (const float4*)(w1 + base);
        const float4* s3 = (const float4*)(w3 + base);
        unsigned* d1 = W1u + ((r * KC + c0) >> 1);
        unsigned* d3 = W3u + ((r * KC + c0) >> 1);
#pragma unroll
        for (int j = 0; j < 8; ++j) {
          float4 v = s1[j];
          d1[2 * j]     = pk2(v.x, v.y);
          d1[2 * j + 1] = pk2(v.z, v.w);
          float4 u = s3[j];
          d3[2 * j]     = pk2(u.x, u.y);
          d3[2 * j + 1] = pk2(u.z, u.w);
        }
        if (kk + KC < HD) {  // pull next k-chunk toward cache (global_prefetch)
          __builtin_prefetch(w1 + base + KC, 0, 1);
          __builtin_prefetch(w3 + base + KC, 0, 1);
        }
      }
      wait_asynccnt0();
      __syncthreads();

#pragma unroll
      for (int ks = 0; ks < KC; ks += 32) {
        const v16bf A = ldsA<KC / 2>(Xu, mt + l16, ks, khalf);
#pragma unroll
        for (int nt = 0; nt < 2; ++nt) {
          const int n = ng + nt * 16 + l16;
          const v16bf B1 = ldsB<KC / 2>(W1u, n, ks, khalf);
          a1[nt] = wmma_bf16(A, B1, a1[nt]);
          const v16bf B3 = ldsB<KC / 2>(W3u, n, ks, khalf);
          a3[nt] = wmma_bf16(A, B3, a3[nt]);
        }
      }
      __syncthreads();
    }

    // silu(h1)*h3 -> ACT (accumulator layout: M = mt + 8*khalf + r, N per lane)
#pragma unroll
    for (int nt = 0; nt < 2; ++nt) {
#pragma unroll
      for (int r = 0; r < 8; ++r) {
        const int m = mt + (khalf << 3) + r;
        const int i = isl + ng + nt * 16 + l16;
        const float h1 = a1[nt][r], h3 = a3[nt][r];
        const float act = (h1 / (1.0f + __expf(-h1))) * h3;
        ACT[m * ID + i] = f2bf(act);
      }
    }
    __syncthreads();
  }

  // ---------------- Phase 2: out += cw * (act @ w2^T), K = ID on-chip ------
  for (int hc = 0; hc < HD; hc += NC) {
    v8f c[2];
    c[0] = zero8; c[1] = zero8;
    for (int ic = 0; ic < ID; ic += NC) {
      {  // stage w2 chunk: 128 h-rows x 128 i-cols fp32 -> bf16
        const int r = tid >> 1, c0 = (tid & 1) * 64;
        const size_t base = ((size_t)e * HD + hc + r) * ID + ic + c0;
        const float4* s = (const float4*)(w2 + base);
        unsigned* d = W2u + ((r * NC + c0) >> 1);
#pragma unroll
        for (int j = 0; j < 16; ++j) {
          float4 v = s[j];
          d[2 * j]     = pk2(v.x, v.y);
          d[2 * j + 1] = pk2(v.z, v.w);
        }
        if (ic + NC < ID) {  // prefetch next i-chunk of w2
          __builtin_prefetch(w2 + base + NC, 0, 1);
        }
      }
      __syncthreads();
#pragma unroll
      for (int nt = 0; nt < 2; ++nt) {
#pragma unroll
        for (int ks = 0; ks < NC; ks += 32) {
          const v16bf A = ldsA<ID / 2>(ACTu, mt + l16, ic + ks, khalf);
          const v16bf B = ldsB<NC / 2>(W2u, ng + nt * 16 + l16, ks, khalf);
          c[nt] = wmma_bf16(A, B, c[nt]);
        }
      }
      __syncthreads();
    }
    // weighted accumulate into out (token appears in 2 expert buckets)
#pragma unroll
    for (int nt = 0; nt < 2; ++nt) {
#pragma unroll
      for (int r = 0; r < 8; ++r) {
        const int m = mt + (khalf << 3) + r;
        const int t = s_idx[m];
        if (t >= 0) {
          const int h = hc + ng + nt * 16 + l16;
          atomicAdd(&out[(size_t)t * HD + h], s_cw[m] * c[nt][r]);
        }
      }
    }
  }
}

// ---------------- host-side launcher ----------------
extern "C" void kernel_launch(void* const* d_in, const int* in_sizes, int n_in,
                              void* d_out, int out_size, void* d_ws,
                              size_t ws_size, hipStream_t stream) {
  const float* x    = (const float*)d_in[0];  // [T, H]
  const float* gw   = (const float*)d_in[1];  // [E, H]
  const float* w1   = (const float*)d_in[2];  // [E, I, H]
  const float* w3   = (const float*)d_in[3];  // [E, I, H]
  const float* w2   = (const float*)d_in[4];  // [E, H, I]
  const float* bias = (const float*)d_in[5];  // [E]
  float* out = (float*)d_out;                 // [T, H] fp32

  // workspace carve (~17 MB)
  char* ws = (char*)d_ws;
  unsigned short* xb = (unsigned short*)ws;                       // 16 MB bf16 x
  float* combine = (float*)(ws + (size_t)NT * HD * 2);            // 128 KB
  int* lists = (int*)(ws + (size_t)NT * HD * 2 + (size_t)NT * NE * 4);  // 128 KB
  int* counts = (int*)(ws + (size_t)NT * HD * 2 + (size_t)NT * NE * 4 +
                       (size_t)NE * NT * 4);

  const int n = NT * HD;
  moe_init_kernel<<<(n + 255) / 256, 256, 0, stream>>>(out, n, counts);
  moe_router_kernel<<<NT, 256, 0, stream>>>(x, gw, bias, xb, combine, counts,
                                            lists);

  dim3 grid(NE, NT / MT);
  const size_t shmem = 102400;  // ACT 64K + X 4K + W1 16K + W3 16K (W2 reuses)
  moe_expert_kernel<<<grid, 256, shmem, stream>>>(w1, w3, w2, xb, combine,
                                                  counts, lists, out);
}
